// sin_NeuralODE_21629455302759
// MI455X (gfx1250) — matrix operand order repaired and checked by reading, not compiled, not run on hardware
//
#include <hip/hip_runtime.h>

#define B_TOT   4096
#define S_LEN   200
#define IN_D    8
#define H_D     128
#define OUT_D   8
#define P_STEPS 50

typedef __attribute__((ext_vector_type(16))) _Float16 v16h;
typedef __attribute__((ext_vector_type(8)))  _Float16 v8h;
typedef __attribute__((ext_vector_type(8)))  float    v8f;

__device__ __forceinline__ float fast_tanh(float x) {
#if __has_builtin(__builtin_amdgcn_tanhf)
    return __builtin_amdgcn_tanhf(x);
#elif __has_builtin(__builtin_amdgcn_tanh_f32)
    return __builtin_amdgcn_tanh_f32(x);
#else
    float e = __builtin_amdgcn_exp2f(x * 2.8853900817779268f);
    return 1.0f - 2.0f * __builtin_amdgcn_rcpf(e + 1.0f);
#endif
}

__global__ __launch_bounds__(256) void neural_ode_kernel(
    const float* __restrict__ x,    const float* __restrict__ Win,
    const float* __restrict__ bin_, const float* __restrict__ W1,
    const float* __restrict__ b1,   const float* __restrict__ W2,
    const float* __restrict__ b2,   const float* __restrict__ Wout,
    const float* __restrict__ bout, float* __restrict__ out)
{
    // LDS: only the two f16 A-operand ping-pong tiles + staged last-x-slice.
    __shared__ _Float16 bufA[16 * H_D];   // 4 KB : current ytmp (GEMM1 / proj A)
    __shared__ _Float16 bufB[16 * H_D];   // 4 KB : tanh(t)      (GEMM2 A)
    __shared__ float    xs[16 * IN_D];

    const int tid  = threadIdx.x;
    const int wid  = tid >> 5;         // wave id 0..7 == N-tile
    const int lane = tid & 31;
    const int nn   = lane & 15;
    const int hi   = lane >> 4;        // lane half (0/1)
    const int hi8  = hi << 3;
    const int col  = wid * 16 + nn;    // global H column owned in C layout
    const int b0   = blockIdx.x * 16;  // first batch row of this WG

    if (tid < 16 * IN_D) {
        int m = tid >> 3, i = tid & 7;
        xs[tid] = x[(size_t)(b0 + m) * S_LEN * IN_D + (size_t)(S_LEN - 1) * IN_D + i];
    }

    // ---- resident f16 B-fragments: W1, W2 (per-wave N-tile) and Wout ------
    // B (32x16, K x N) layout per lane: column nn; half j<8 <-> K = k*32 + hi*8 + j, j>=8 <-> K = k*32 + 16 + hi*8 + (j-8)
    v16h B1f[4], B2f[4], Wf[4];
    {
        const float* r1 = W1   + (size_t)col       * H_D;
        const float* r2 = W2   + (size_t)col       * H_D;
        const float* rw = Wout + (size_t)(nn & 7)  * H_D;   // Wout is (OUT=8, H)
        #pragma unroll
        for (int k = 0; k < 4; ++k) {
            #pragma unroll
            for (int j = 0; j < 16; ++j) {
                int kk = k * 32 + (j < 8 ? hi * 8 + j : 16 + hi * 8 + (j - 8));
                B1f[k][j] = (_Float16)r1[kk];
                B2f[k][j] = (_Float16)r2[kk];
                Wf [k][j] = (nn < OUT_D) ? (_Float16)rw[kk] : (_Float16)0.0f;
            }
        }
    }
    const float b1v  = b1[col];
    const float b2v  = b2[col];
    const float binv = bin_[col];
    const float outb = (nn < OUT_D) ? bout[nn] : 0.0f;
    float wv[IN_D];
    #pragma unroll
    for (int i = 0; i < IN_D; ++i) wv[i] = Win[(size_t)col * IN_D + i];

    __syncthreads();

    // ---- h0 = x_last @ Win^T + bin, directly in C layout ------------------
    // C layout: slot r -> row m = hi*8 + r, column = col
    float y[8];
    #pragma unroll
    for (int r = 0; r < 8; ++r) {
        int m = hi * 8 + r;
        float s = binv;
        #pragma unroll
        for (int i = 0; i < IN_D; ++i) s += xs[m * IN_D + i] * wv[i];
        y[r] = s;
    }

    // ---- K=128 GEMM: all 8 A-fragment LDS loads issued up front, then the
    //      4-WMMA accumulation chain; bias pre-loaded into the accumulator.
    // A (16x32 f16) layout: lane row = lane&15;
    //   slots 0..7 : K = k*32 + hi8 + j ; slots 8..15 : K = k*32 + 16 + hi8 + j
    auto gemm = [&](const _Float16* buf, const v16h* Bf, float initv) -> v8f {
        const _Float16* base = buf + (lane & 15) * H_D + hi8;
        v8h lo[4], hh[4];
        #pragma unroll
        for (int k = 0; k < 4; ++k) {
            lo[k] = *reinterpret_cast<const v8h*>(base + k * 32);
            hh[k] = *reinterpret_cast<const v8h*>(base + k * 32 + 16);
        }
        v8f acc = {initv, initv, initv, initv, initv, initv, initv, initv};
        #pragma unroll
        for (int k = 0; k < 4; ++k) {
            v16h a;
            #pragma unroll
            for (int j = 0; j < 8; ++j) { a[j] = lo[k][j]; a[j + 8] = hh[k][j]; }
            acc = __builtin_amdgcn_wmma_f32_16x16x32_f16(
                false, a, false, Bf[k], (short)0, acc, false, false);
        }
        return acc;
    };

    auto publishA = [&](const float* v) {      // write ytmp (f16) to bufA
        #pragma unroll
        for (int r = 0; r < 8; ++r)
            bufA[(hi * 8 + r) * H_D + col] = (_Float16)v[r];
    };

    // one ode_f evaluation: bufA -> GEMM1(+b1) -> tanh -> bufB -> GEMM2(+b2)
    auto ode_eval = [&](float* kout) {
        v8f t = gemm(bufA, B1f, b1v);
        #pragma unroll
        for (int r = 0; r < 8; ++r)
            bufB[(hi * 8 + r) * H_D + col] = (_Float16)fast_tanh(t[r]);
        __syncthreads();                       // bufB ready / bufA reads done
        v8f k2 = gemm(bufB, B2f, b2v);
        #pragma unroll
        for (int r = 0; r < 8; ++r) kout[r] = k2[r];
    };

    // ---- RK4 scan; per-step output projection is a WMMA on wave 0 ---------
    float acc_[8], ytmp[8], k[8];
    for (int p = 0; p < P_STEPS; ++p) {
        publishA(y);
        __syncthreads();                       // bufA = y_p, visible to all

        // out[b, p, :] = y_p @ Wout^T + bout : one 16x16x128 WMMA tile
        if (wid == 0) {
            v8f o = gemm(bufA, Wf, outb);
            if (nn < OUT_D) {
                #pragma unroll
                for (int r = 0; r < 8; ++r)
                    out[(size_t)(b0 + hi * 8 + r) * (P_STEPS * OUT_D)
                        + (size_t)p * OUT_D + nn] = o[r];
            }
        }
        if (p == P_STEPS - 1) break;           // uniform exit

        // k1
        ode_eval(k);
        #pragma unroll
        for (int r = 0; r < 8; ++r) {
            acc_[r] = y[r] + k[r] * (1.0f / 6.0f);
            ytmp[r] = y[r] + 0.5f * k[r];
        }
        publishA(ytmp); __syncthreads();
        // k2
        ode_eval(k);
        #pragma unroll
        for (int r = 0; r < 8; ++r) {
            acc_[r] += k[r] * (1.0f / 3.0f);
            ytmp[r] = y[r] + 0.5f * k[r];
        }
        publishA(ytmp); __syncthreads();
        // k3
        ode_eval(k);
        #pragma unroll
        for (int r = 0; r < 8; ++r) {
            acc_[r] += k[r] * (1.0f / 3.0f);
            ytmp[r] = y[r] + k[r];
        }
        publishA(ytmp); __syncthreads();
        // k4
        ode_eval(k);
        #pragma unroll
        for (int r = 0; r < 8; ++r) y[r] = acc_[r] + k[r] * (1.0f / 6.0f);
    }
}

extern "C" void kernel_launch(void* const* d_in, const int* in_sizes, int n_in,
                              void* d_out, int out_size, void* d_ws, size_t ws_size,
                              hipStream_t stream) {
    (void)in_sizes; (void)n_in; (void)out_size; (void)d_ws; (void)ws_size;
    const float* x    = (const float*)d_in[0];
    const float* Win  = (const float*)d_in[1];
    const float* bin_ = (const float*)d_in[2];
    const float* W1   = (const float*)d_in[3];
    const float* b1   = (const float*)d_in[4];
    const float* W2   = (const float*)d_in[5];
    const float* b2   = (const float*)d_in[6];
    const float* Wout = (const float*)d_in[7];
    const float* bout = (const float*)d_in[8];

    dim3 grid(B_TOT / 16);   // 256 workgroups x 16 batch rows
    dim3 block(256);         // 8 waves, one N-tile each
    neural_ode_kernel<<<grid, block, 0, stream>>>(
        x, Win, bin_, W1, b1, W2, b2, Wout, bout, (float*)d_out);
}
